// Bert_Proj_CRF_73959336837468
// MI455X (gfx1250) — matrix operand order repaired and checked by reading, not compiled, not run on hardware
//
#include <hip/hip_runtime.h>
#include <hip/hip_bf16.h>

// ---------------------------------------------------------------------------
// Bert_Proj_CRF for MI455X (gfx1250, wave32, WMMA, async LDS copy)
//   Phase 0: one-time f32 -> bf16 conversion of embed_w / fc1_w / dom_w (ws)
//   Phase 1: fused embed-gather + [x|dom] @ fc1^T (bf16 WMMA, f32 acc) + relu
//            + fc2 + log_softmax -> logits [B*S, 8] (f32, in ws)
//            fc1 weights double-buffered in LDS via GLOBAL_LOAD_ASYNC_TO_LDS
//            (runtime-toggled buffer offset, loop kept un-unrolled so the
//             accumulators stay pinned -> no register-rotation movs)
//   Phase 2: CRF forward scan + gold path score -> loss [B]
// ---------------------------------------------------------------------------

typedef __attribute__((ext_vector_type(16))) __bf16 v16bf;
typedef __attribute__((ext_vector_type(8)))  float  v8f;
typedef __attribute__((ext_vector_type(4)))  int    v4i_t;

#define BATCH   64
#define SEQ     512
#define EMB     768
#define HID     256
#define KDIM    1024   // EMB + HID
#define NTAGS   8
#define HPAD    260    // padded Hsh row stride (floats), 16B aligned, bank-spread
#define BUFBYTES 16384 // one B-tile buffer: 256 n x 32 k bf16

#if __has_builtin(__builtin_amdgcn_global_load_async_to_lds_b128)
#define USE_ASYNC_LDS 1
typedef __attribute__((address_space(1))) v4i_t gv4i_t;   // global-AS 4xi32
typedef __attribute__((address_space(3))) v4i_t lv4i_t;   // LDS-AS 4xi32
#else
#define USE_ASYNC_LDS 0
#endif

__device__ __forceinline__ void async_wait_all() {
#if USE_ASYNC_LDS
#if __has_builtin(__builtin_amdgcn_s_wait_asynccnt)
    __builtin_amdgcn_s_wait_asynccnt(0);
#else
    asm volatile("s_wait_asynccnt 0x0" ::: "memory");
#endif
#endif
}

__device__ __forceinline__ unsigned short f2bf(float x) {
    unsigned u = __float_as_uint(x);
    u += 0x7FFFu + ((u >> 16) & 1u);        // round-to-nearest-even
    return (unsigned short)(u >> 16);
}
__device__ __forceinline__ unsigned pack2(float lo, float hi) {
    return (unsigned)f2bf(lo) | ((unsigned)f2bf(hi) << 16);
}

union ABu {
    v16bf v;
    unsigned short us[16];
    uint4 q[2];
};

// ---------------------------------------------------------------------------
// Phase 0: pack f32 -> bf16 (pairs), 8 floats (one uint4 out) per thread.
// ---------------------------------------------------------------------------
extern "C" __global__ __launch_bounds__(256)
void cvt_f32_bf16_kernel(const float* __restrict__ src,
                         uint4* __restrict__ dst, int n8)
{
    const int i = blockIdx.x * 256 + threadIdx.x;
    if (i < n8) {
        float4 a = ((const float4*)src)[i * 2 + 0];
        float4 b = ((const float4*)src)[i * 2 + 1];
        uint4 p;
        p.x = pack2(a.x, a.y); p.y = pack2(a.z, a.w);
        p.z = pack2(b.x, b.y); p.w = pack2(b.z, b.w);
        dst[i] = p;
    }
}

// ---------------------------------------------------------------------------
// Phase 1: fused GEMM.  grid = 512 blocks (64 rows each), block = 512 thr
// (16 waves: 4 M-subtiles x 4 N-quarters -> only 4 v8f accumulators/wave).
// LDS: Bsh = fc1 K-chunk bf16 [2 buf][n=256][k=32]  (2 x 16 KB)
//      Hsh = relu(fc1) tile f32 [row=64][HPAD]       (~65 KB)
// ---------------------------------------------------------------------------
extern "C" __global__ __launch_bounds__(512)
void fused_embed_mlp_kernel(const int*            __restrict__ words,
                            const int*            __restrict__ corpus,
                            const unsigned short* __restrict__ embed_bf,
                            const unsigned short* __restrict__ dom_bf,
                            const uint4*          __restrict__ fc1_bfq,  // [256][128] uint4
                            const float*          __restrict__ fc1_b,
                            const float*          __restrict__ fc2_w,
                            const float*          __restrict__ fc2_b,
                            float*                __restrict__ logits)
{
    extern __shared__ char smem[];
    float* Hsh = (float*)(smem + 2 * BUFBYTES);   // 64 * HPAD floats

    const int tid  = threadIdx.x;
    const int lane = tid & 31;
    const int wave = tid >> 5;                    // 0..15
    const int mi   = wave & 3;                    // M subtile (16 rows)
    const int nq   = wave >> 2;                   // N quarter (64 cols)

    const int m0   = blockIdx.x * 64;
    const int rloc = mi * 16 + (lane & 15);
    const int m    = m0 + rloc;

    const int word = words[m];
    const int bidx = m >> 9;                      // m / SEQ
    const unsigned short* erow = embed_bf + (size_t)word * EMB;
    const unsigned short* drow = dom_bf  + (size_t)corpus[bidx] * HID;

    // A-fragment (16-bit A 16x32): lanes 0-15 -> K 0..15,
    // lanes 16-31 -> K 16..31.
    const int koA = (lane >= 16) ? 16 : 0;
    // B-fragment (32x16): lanes 0-15 -> K 0..15, lanes 16-31 -> K 16..31.
    const int koB = (lane >= 16) ? 16 : 0;

    v8f acc[4];
    {
        v8f vz = {0.f,0.f,0.f,0.f,0.f,0.f,0.f,0.f};
        #pragma unroll
        for (int t = 0; t < 4; ++t) acc[t] = vz;
    }

    // byte offset of this thread's piece inside one B buffer
    const int nrow    = tid >> 1;
    const int half    = tid & 1;
    const int stageLo = nrow * 64 + half * 32;                // bytes in buffer
    // this wave's B-fragment base byte offset inside one B buffer
    const int fragLo  = (nq * 64 + (lane & 15)) * 64 + koB * 2;

    // stage one 16KB K-chunk of fc1 bf16 weights into LDS buffer at `off`
    auto stage = [&](int ks, unsigned off) {
        const uint4* gsrc = fc1_bfq + (size_t)nrow * 128 + ks * 4 + half * 2;
#if USE_ASYNC_LDS
        gv4i_t* gp = (gv4i_t*)gsrc;
        lv4i_t* lp = (lv4i_t*)(smem + off + stageLo);
        __builtin_amdgcn_global_load_async_to_lds_b128(gp, lp, 0,  0);
        __builtin_amdgcn_global_load_async_to_lds_b128(gp, lp, 16, 0);
#else
        uint4* lp = (uint4*)(smem + off + stageLo);
        const uint4* gs = gsrc;
        lp[0] = gs[0];
        lp[1] = gs[1];
#endif
    };

    // ---- prologue: fill buffer 0 ----
    stage(0, 0);
    async_wait_all();
    __syncthreads();

    unsigned bufoff = 0;
    #pragma unroll 1
    for (int ks = 0; ks < 32; ++ks) {
        if (ks + 1 < 32) stage(ks + 1, bufoff ^ BUFBYTES);

        // ---- A fragment: bf16 gather straight from L2-resident tables ----
        const int k0 = ks * 32;
        const unsigned short* src = (k0 < EMB) ? (erow + k0) : (drow + (k0 - EMB));
        ABu A;
        A.q[0] = *(const uint4*)(src + koA);
        A.q[1] = *(const uint4*)(src + koA + 8);

        // ---- 4 WMMA tiles across this wave's 64 output cols ----
        const char* bbase = smem + bufoff + fragLo;
        #pragma unroll
        for (int t = 0; t < 4; ++t) {
            ABu B;
            B.q[0] = *(const uint4*)(bbase + t * 1024);
            B.q[1] = *(const uint4*)(bbase + t * 1024 + 16);
            acc[t] = __builtin_amdgcn_wmma_f32_16x16x32_bf16(
                         false, A.v, false, B.v, (short)0, acc[t], false, false);
        }

        async_wait_all();
        __syncthreads();
        bufoff ^= BUFBYTES;
    }

    // ---- epilogue: bias + relu -> Hsh ----
    {
        const int rbase = mi * 16 + ((lane >= 16) ? 8 : 0);
        #pragma unroll
        for (int t = 0; t < 4; ++t) {
            const int n = nq * 64 + t * 16 + (lane & 15);
            const float bias = fc1_b[n];
            #pragma unroll
            for (int v = 0; v < 8; ++v) {
                float val = acc[t][v] + bias;
                Hsh[(rbase + v) * HPAD + n] = fmaxf(val, 0.f);
            }
        }
    }
    __syncthreads();

    // ---- fc2 (256 -> 8) + log_softmax, one thread per row ----
    if (tid < 64) {
        const int row = tid;
        const float* h = Hsh + row * HPAD;
        float pred[NTAGS];
        #pragma unroll
        for (int t = 0; t < NTAGS; ++t) {
            const float* w = fc2_w + t * HID;
            float4 a4 = {0.f, 0.f, 0.f, 0.f};
            for (int n = 0; n < HID; n += 4) {
                float4 h4 = *(const float4*)(h + n);
                float4 w4 = *(const float4*)(w + n);
                a4.x += h4.x * w4.x; a4.y += h4.y * w4.y;
                a4.z += h4.z * w4.z; a4.w += h4.w * w4.w;
            }
            pred[t] = fc2_b[t] + (a4.x + a4.y) + (a4.z + a4.w);
        }
        float mx = pred[0];
        #pragma unroll
        for (int t = 1; t < NTAGS; ++t) mx = fmaxf(mx, pred[t]);
        float se = 0.f;
        #pragma unroll
        for (int t = 0; t < NTAGS; ++t) se += __expf(pred[t] - mx);
        const float lse = mx + __logf(se);
        float* outp = logits + (size_t)(m0 + row) * NTAGS;
        #pragma unroll
        for (int t = 0; t < NTAGS; ++t) outp[t] = pred[t] - lse;
    }
}

// ---------------------------------------------------------------------------
// Phase 2: CRF loss.  grid = 64 (one wave per batch element), block = 32.
// ---------------------------------------------------------------------------
extern "C" __global__ __launch_bounds__(32)
void crf_loss_kernel(const float* __restrict__ logits,
                     const int*   __restrict__ words,
                     const int*   __restrict__ target,
                     const float* __restrict__ trans,
                     const float* __restrict__ start_s,
                     const float* __restrict__ end_s,
                     float*       __restrict__ out)
{
    const int b    = blockIdx.x;
    const int lane = threadIdx.x;
    const int j    = lane & 7;                      // tag index for this lane

    const float* L = logits + (size_t)b * SEQ * NTAGS;
    const int*   W = words  + b * SEQ;
    const int*   T = target + b * SEQ;

    float tcol[NTAGS];                              // trans[i][j] column
    #pragma unroll
    for (int i = 0; i < NTAGS; ++i) tcol[i] = trans[i * NTAGS + j];

    // ---- forward scan ----
    float alpha = L[j] + start_s[j];
    for (int s = 1; s < SEQ; ++s) {
        const float emit = L[s * NTAGS + j];
        const bool  msk  = (W[s] != 0);
        float av[NTAGS];
        float mv = -3.4e38f;
        #pragma unroll
        for (int i = 0; i < NTAGS; ++i) {
            av[i] = __shfl(alpha, i, 32) + tcol[i];
            mv = fmaxf(mv, av[i]);
        }
        float ss = 0.f;
        #pragma unroll
        for (int i = 0; i < NTAGS; ++i) ss += __expf(av[i] - mv);
        const float nw = mv + __logf(ss) + emit;
        alpha = msk ? nw : alpha;
    }

    // ---- log partition: logsumexp over tags (8-lane groups) ----
    float val = alpha + end_s[j];
    float mv = val;
    mv = fmaxf(mv, __shfl_xor(mv, 1, 32));
    mv = fmaxf(mv, __shfl_xor(mv, 2, 32));
    mv = fmaxf(mv, __shfl_xor(mv, 4, 32));
    float se = __expf(val - mv);
    se += __shfl_xor(se, 1, 32);
    se += __shfl_xor(se, 2, 32);
    se += __shfl_xor(se, 4, 32);
    const float norm = mv + __logf(se);

    // ---- gold path score (parallel over s) ----
    float es = 0.f, ts = 0.f;
    int cnt = 0;
    for (int s = lane; s < SEQ; s += 32) {
        const int msk = (W[s] != 0);
        const int tg  = T[s];
        if (msk) es += L[s * NTAGS + tg];
        cnt += msk;
        if (s > 0 && msk) ts += trans[T[s - 1] * NTAGS + tg];
    }
    float tot = es + ts;
    tot += __shfl_xor(tot, 1, 32);  cnt += __shfl_xor(cnt, 1, 32);
    tot += __shfl_xor(tot, 2, 32);  cnt += __shfl_xor(cnt, 2, 32);
    tot += __shfl_xor(tot, 4, 32);  cnt += __shfl_xor(cnt, 4, 32);
    tot += __shfl_xor(tot, 8, 32);  cnt += __shfl_xor(cnt, 8, 32);
    tot += __shfl_xor(tot, 16, 32); cnt += __shfl_xor(cnt, 16, 32);

    if (lane == 0) {
        int last = cnt - 1;
        if (last < 0) last = 0;
        const float score = tot + start_s[T[0]] + end_s[T[last]];
        out[b] = norm - score;
    }
}

// ---------------------------------------------------------------------------
extern "C" void kernel_launch(void* const* d_in, const int* in_sizes, int n_in,
                              void* d_out, int out_size, void* d_ws, size_t ws_size,
                              hipStream_t stream) {
    const int*   words   = (const int*)  d_in[0];
    const int*   target  = (const int*)  d_in[1];
    const int*   corpus  = (const int*)  d_in[2];
    const float* embed_w = (const float*)d_in[3];
    const float* dom_w   = (const float*)d_in[4];
    const float* fc1_w   = (const float*)d_in[5];
    const float* fc1_b   = (const float*)d_in[6];
    const float* fc2_w   = (const float*)d_in[7];
    const float* fc2_b   = (const float*)d_in[8];
    const float* trans   = (const float*)d_in[9];
    const float* start_s = (const float*)d_in[10];
    const float* end_s   = (const float*)d_in[11];

    float* out = (float*)d_out;                     // [64]

    // ---- workspace layout ----
    char* ws = (char*)d_ws;
    float*          logits   = (float*)ws;                        //  1,048,576 B
    unsigned short* embed_bf = (unsigned short*)(ws + (1 << 20)); // 32,452,608 B
    unsigned short* fc1_bf   = (unsigned short*)((char*)embed_bf + 32452608);
    unsigned short* dom_bf   = (unsigned short*)((char*)fc1_bf + 524288);

    // ---- Phase 0: one-time bf16 conversions ----
    {
        const int n8_embed = (21128 * EMB) / 8;   // 2,028,288
        const int n8_fc1   = (HID * KDIM) / 8;    // 32,768
        const int n8_dom   = (10 * HID) / 8;      // 320
        cvt_f32_bf16_kernel<<<(n8_embed + 255) / 256, 256, 0, stream>>>(
            embed_w, (uint4*)embed_bf, n8_embed);
        cvt_f32_bf16_kernel<<<(n8_fc1 + 255) / 256, 256, 0, stream>>>(
            fc1_w, (uint4*)fc1_bf, n8_fc1);
        cvt_f32_bf16_kernel<<<(n8_dom + 255) / 256, 256, 0, stream>>>(
            dom_w, (uint4*)dom_bf, n8_dom);
    }

    // ---- Phase 1: fused embed + MLP + log_softmax ----
    const size_t shmem = 2 * BUFBYTES + (size_t)64 * HPAD * sizeof(float);
    fused_embed_mlp_kernel<<<(BATCH * SEQ) / 64, 512, shmem, stream>>>(
        words, corpus, embed_bf, dom_bf, (const uint4*)fc1_bf,
        fc1_b, fc2_w, fc2_b, logits);

    // ---- Phase 2: CRF ----
    crf_loss_kernel<<<BATCH, 32, 0, stream>>>(
        logits, words, target, trans, start_s, end_s, out);
}
